// SimpleRoPEWithBias_7739531067590
// MI455X (gfx1250) — matrix operand order repaired and checked by reading, not compiled, not run on hardware
//
#include <hip/hip_runtime.h>
#include <math.h>

// ---------------- problem constants ----------------
#define B_   2
#define S_   2048
#define H_   1024
#define NH_  16
#define HD_  64           // H_/NH_
// ln(10000)/32
#define LOG1E4_OVER_32 0.28782313662425576f

typedef _Float16 half_t;
typedef __attribute__((ext_vector_type(16))) _Float16 v16h;
typedef __attribute__((ext_vector_type(8)))  _Float16 v8h;
typedef __attribute__((ext_vector_type(4)))  _Float16 v4h;
typedef __attribute__((ext_vector_type(8)))  float    v8f;
typedef __attribute__((ext_vector_type(4)))  float    v4f;

union HV16 { v16h v; v8h h[2]; _Float16 e[16]; };

// ---------------- WMMA helpers ----------------
// D = A(16x32 f16) x B(32x16 f16) + C(16x16 f32)
__device__ __forceinline__ v8f wmma_f16(v16h a, v16h b, v8f c) {
  return __builtin_amdgcn_wmma_f32_16x16x32_f16(false, a, false, b, (short)0, c,
                                                false, false);
}

// A fragment: row-major A[m][k], tile at (m0,k0). Per ISA 16-bit A layout:
// lane<16 : M=lane,    e0..7 -> K=k0+e,    e8..15 -> K=k0+16+(e-8)
// lane>=16: M=lane-16, e0..7 -> K=k0+8+e,  e8..15 -> K=k0+24+(e-8)
__device__ __forceinline__ v16h load_a_frag(const half_t* __restrict__ A, int lda,
                                            int m0, int k0, int lane) {
  int row = m0 + (lane & 15);
  int kb  = k0 + ((lane >> 4) << 3);
  const half_t* p = A + (size_t)row * lda + kb;
  HV16 r;
  r.h[0] = *reinterpret_cast<const v8h*>(p);
  r.h[1] = *reinterpret_cast<const v8h*>(p + 16);
  return r.v;
}

// Same A fragment but converting fp32 source on the fly (for attn matrix).
__device__ __forceinline__ v16h load_a_frag_f32(const float* __restrict__ A, int lda,
                                                int m0, int k0, int lane) {
  int row = m0 + (lane & 15);
  int kb  = k0 + ((lane >> 4) << 3);
  const float* p0 = A + (size_t)row * lda + kb;
  const float* p1 = p0 + 16;
  HV16 r;
#pragma unroll
  for (int e = 0; e < 8; ++e) {
    r.e[e]     = (_Float16)p0[e];
    r.e[8 + e] = (_Float16)p1[e];
  }
  return r.v;
}

// B fragment for D = A x W^T: B[k][n] = W[n][k], W row-major [n][ldb].
// lane holds column N = n0 + (lane&15); e0..7 -> K = k0 + (lane>=16)*8 + e,
// e8..15 -> K = k0 + 16 + (lane>=16)*8 + (e-8) (same order as A fragment).
__device__ __forceinline__ v16h load_b_frag(const half_t* __restrict__ W, int ldb,
                                            int n0, int k0, int lane) {
  int nn = n0 + (lane & 15);
  int kb = k0 + ((lane >> 4) << 3);
  const half_t* p = W + (size_t)nn * ldb + kb;
  HV16 r;
  r.h[0] = *reinterpret_cast<const v8h*>(p);
  r.h[1] = *reinterpret_cast<const v8h*>(p + 16);
  return r.v;
}

// ---------------- kernel 0: fp32 -> f16 convert ----------------
__global__ __launch_bounds__(256) void k_cvt(const float* __restrict__ s,
                                             half_t* __restrict__ d, int n4) {
  int i = blockIdx.x * 256 + threadIdx.x;
  if (i < n4) {
    v4f x = *reinterpret_cast<const v4f*>(s + (size_t)i * 4);
    v4h y;
#pragma unroll
    for (int e = 0; e < 4; ++e) y[e] = (_Float16)x[e];
    *reinterpret_cast<v4h*>(d + (size_t)i * 4) = y;
  }
}

// ---------------- kernel 1: fused QKV projection + RoPE ----------------
// grid: (M/16 = 256, NH_ = 16, 3 {q,k,v}); block: 32 (one wave).
// Each wave: 16 rows x full head width (64) => 4 accumulators, K-loop over H.
__global__ __launch_bounds__(32) void k_qkv_rope(
    const half_t* __restrict__ xh, const half_t* __restrict__ Wqkv,
    half_t* __restrict__ qh, half_t* __restrict__ kh, half_t* __restrict__ vTh) {
  const int lane  = threadIdx.x & 31;
  const int mtile = blockIdx.x;
  const int head  = blockIdx.y;
  const int which = blockIdx.z;             // 0=q 1=k 2=v
  const half_t* W = Wqkv + (size_t)which * H_ * H_;
  const int n0 = head * HD_;

  v8f acc[4] = {};
  for (int k0 = 0; k0 < H_; k0 += 32) {
    v16h a = load_a_frag(xh, H_, mtile * 16, k0, lane);
#pragma unroll
    for (int t = 0; t < 4; ++t)
      acc[t] = wmma_f16(a, load_b_frag(W, H_, n0 + 16 * t, k0, lane), acc[t]);
  }

  const int rhalf  = lane >> 4;             // 0/1 -> +8 rows
  const int ncol   = lane & 15;
  const int b      = (mtile * 16) / S_;     // same batch for whole tile
  const int sbase0 = (mtile * 16) % S_;

  if (which < 2) {                          // RoPE on q and k, in registers
    float res[4][8];
#pragma unroll
    for (int t = 0; t < 4; ++t) {
      float invf = expf(-(float)((ncol + 16 * t) & 31) * LOG1E4_OVER_32);
#pragma unroll
      for (int i = 0; i < 8; ++i) {
        float ang = (float)(sbase0 + 8 * rhalf + i) * invf;
        float sn, cs;
        sincosf(ang, &sn, &cs);
        float rot = (t < 2) ? -acc[t + 2][i] : acc[t - 2][i];
        res[t][i] = acc[t][i] * cs + rot * sn;
      }
    }
#pragma unroll
    for (int t = 0; t < 4; ++t)
#pragma unroll
      for (int i = 0; i < 8; ++i) acc[t][i] = res[t][i];
  }

  // stage 16x64 f16 tile in LDS for coalesced stores
  __shared__ alignas(16) _Float16 lt[16 * 64];
#pragma unroll
  for (int t = 0; t < 4; ++t)
#pragma unroll
    for (int i = 0; i < 8; ++i)
      lt[(i + 8 * rhalf) * 64 + ncol + 16 * t] = (_Float16)acc[t][i];
  __syncthreads();

  if (which < 2) {
    // q/k: [b, head, s, d] row-major, rows of 64 halfs
    half_t* dstb = (which == 0 ? qh : kh) +
                   ((size_t)(b * NH_ + head) * S_ + sbase0) * HD_;
#pragma unroll
    for (int j = 0; j < 4; ++j) {
      int c = lane + 32 * j;                // 128 chunks of 8 halfs
      int r = c >> 3, off = (c & 7) * 8;
      *reinterpret_cast<v8h*>(dstb + (size_t)r * HD_ + off) =
          *reinterpret_cast<const v8h*>(&lt[r * 64 + off]);
    }
  } else {
    // v transposed: [b, head, d, s]
    half_t* dstb = vTh + (size_t)(b * NH_ + head) * HD_ * S_ + sbase0;
#pragma unroll
    for (int j = 0; j < 2; ++j) {
      int d = lane + 32 * j;
      alignas(16) _Float16 tmp[16];
#pragma unroll
      for (int r = 0; r < 16; ++r) tmp[r] = lt[r * 64 + d];
      half_t* dp = dstb + (size_t)d * S_;
      *reinterpret_cast<v8h*>(dp)     = *reinterpret_cast<const v8h*>(&tmp[0]);
      *reinterpret_cast<v8h*>(dp + 8) = *reinterpret_cast<const v8h*>(&tmp[8]);
    }
  }
}

// ---------------- kernel 2a: scores = qk^T/8 + bias, causal ----------------
// grid: (S/16 = 128 q-tiles, B*NH = 32); block: 256 (8 waves, strided k-tiles).
// Only lower-triangle tiles are computed/stored.
__global__ __launch_bounds__(256) void k_scores(
    const half_t* __restrict__ qh, const half_t* __restrict__ kh,
    const float* __restrict__ bias, float* __restrict__ attn) {
  const int lane = threadIdx.x & 31;
  const int wave = threadIdx.x >> 5;
  const int qt   = blockIdx.x;
  const int bh   = blockIdx.y;
  const half_t* qp = qh + (size_t)bh * S_ * HD_;
  const half_t* kp = kh + (size_t)bh * S_ * HD_;
  float* attnp = attn + (size_t)bh * S_ * S_;

  v16h a0 = load_a_frag(qp, HD_, qt * 16, 0, lane);
  v16h a1 = load_a_frag(qp, HD_, qt * 16, 32, lane);

  __shared__ alignas(16) float st[8][16 * 16];
  const int rhalf = lane >> 4, ncol = lane & 15;

  const int ntk   = qt + 1;                 // valid k-tiles (causal)
  const int iters = (ntk + 7) >> 3;
  for (int it = 0; it < iters; ++it) {
    int  kt  = wave + it * 8;
    bool act = kt < ntk;
    if (act) {
      v8f acc = {};
      acc = wmma_f16(a0, load_b_frag(kp, HD_, kt * 16, 0, lane), acc);
      acc = wmma_f16(a1, load_b_frag(kp, HD_, kt * 16, 32, lane), acc);
#pragma unroll
      for (int i = 0; i < 8; ++i) {
        int sq = qt * 16 + i + 8 * rhalf;
        int sk = kt * 16 + ncol;
        float v = acc[i] * 0.125f + bias[(size_t)sq * S_ + sk];
        if (sk > sq) v = -__builtin_inff();
        st[wave][(i + 8 * rhalf) * 16 + ncol] = v;
      }
    }
    __syncthreads();
    if (act) {
#pragma unroll
      for (int j = 0; j < 2; ++j) {
        int c = lane + 32 * j;              // 64 chunks of 4 floats
        int r = c >> 2, off = (c & 3) * 4;
        *reinterpret_cast<v4f*>(attnp + (size_t)(qt * 16 + r) * S_ + kt * 16 + off) =
            *reinterpret_cast<const v4f*>(&st[wave][r * 16 + off]);
      }
    }
    __syncthreads();
  }
}

// ---------------- kernel 2b: causal row softmax (in place) ----------------
// grid: B*NH*S rows; block 256; each thread holds 8 contiguous elements.
__global__ __launch_bounds__(256) void k_softmax(float* __restrict__ attn) {
  const size_t row = blockIdx.x;
  const int sq    = (int)(row & (S_ - 1));
  const int valid = sq + 1;
  float* p = attn + row * S_;
  const int tid  = threadIdx.x;
  const int base = tid * 8;

  float v[8];
  {
    v4f x0 = *reinterpret_cast<const v4f*>(p + base);
    v4f x1 = *reinterpret_cast<const v4f*>(p + base + 4);
#pragma unroll
    for (int e = 0; e < 4; ++e) { v[e] = x0[e]; v[4 + e] = x1[e]; }
  }
  float m = -__builtin_inff();
#pragma unroll
  for (int e = 0; e < 8; ++e) {
    if (base + e >= valid) v[e] = -__builtin_inff();
    m = fmaxf(m, v[e]);
  }
  __shared__ float red[256];
  red[tid] = m;
  __syncthreads();
  for (int s2 = 128; s2 > 0; s2 >>= 1) {
    if (tid < s2) red[tid] = fmaxf(red[tid], red[tid + s2]);
    __syncthreads();
  }
  float mx = red[0];
  __syncthreads();

  float sum = 0.f;
#pragma unroll
  for (int e = 0; e < 8; ++e) {
    v[e] = (base + e < valid) ? expf(v[e] - mx) : 0.f;
    sum += v[e];
  }
  red[tid] = sum;
  __syncthreads();
  for (int s2 = 128; s2 > 0; s2 >>= 1) {
    if (tid < s2) red[tid] += red[tid + s2];
    __syncthreads();
  }
  float inv = 1.0f / red[0];

  v4f y0, y1;
#pragma unroll
  for (int e = 0; e < 4; ++e) { y0[e] = v[e] * inv; y1[e] = v[4 + e] * inv; }
  *reinterpret_cast<v4f*>(p + base)     = y0;
  *reinterpret_cast<v4f*>(p + base + 4) = y1;
}

// ---------------- kernel 2c: o = attn @ v  (causal tile skipping) ----------
// grid: (128 q-tiles, 32 bh); block 32. Output f16 [b, s, head*64+d].
__global__ __launch_bounds__(32) void k_av(const float* __restrict__ attn,
                                           const half_t* __restrict__ vTh,
                                           half_t* __restrict__ oh) {
  const int lane = threadIdx.x & 31;
  const int qt = blockIdx.x, bh = blockIdx.y;
  const int b = bh / NH_, head = bh % NH_;
  const float*  ap = attn + (size_t)bh * S_ * S_;
  const half_t* vp = vTh + (size_t)bh * HD_ * S_;

  v8f acc[4] = {};
  const int kmax = qt * 16 + 15;
  for (int k0 = 0; k0 <= kmax; k0 += 32) {
    v16h a = load_a_frag_f32(ap, S_, qt * 16, k0, lane);
#pragma unroll
    for (int t = 0; t < 4; ++t)
      acc[t] = wmma_f16(a, load_b_frag(vp, S_, 16 * t, k0, lane), acc[t]);
  }

  __shared__ alignas(16) _Float16 lt[16 * 64];
  const int rhalf = lane >> 4, ncol = lane & 15;
#pragma unroll
  for (int t = 0; t < 4; ++t)
#pragma unroll
    for (int i = 0; i < 8; ++i)
      lt[(i + 8 * rhalf) * 64 + ncol + 16 * t] = (_Float16)acc[t][i];
  __syncthreads();

  half_t* dstb = oh + (size_t)(b * S_ + qt * 16) * H_ + head * HD_;
#pragma unroll
  for (int j = 0; j < 4; ++j) {
    int c = lane + 32 * j;
    int r = c >> 3, off = (c & 7) * 8;
    *reinterpret_cast<v8h*>(dstb + (size_t)r * H_ + off) =
        *reinterpret_cast<const v8h*>(&lt[r * 64 + off]);
  }
}

// ---------------- kernel 3: final = o @ Wo^T ----------------
// grid: (256 m-tiles, 16 n-groups of 64); block 32. fp32 output.
__global__ __launch_bounds__(32) void k_outproj(const half_t* __restrict__ oh,
                                                const half_t* __restrict__ Woh,
                                                float* __restrict__ outp) {
  const int lane  = threadIdx.x & 31;
  const int mtile = blockIdx.x;
  const int n0    = blockIdx.y * 64;

  v8f acc[4] = {};
  for (int k0 = 0; k0 < H_; k0 += 32) {
    v16h a = load_a_frag(oh, H_, mtile * 16, k0, lane);
#pragma unroll
    for (int t = 0; t < 4; ++t)
      acc[t] = wmma_f16(a, load_b_frag(Woh, H_, n0 + 16 * t, k0, lane), acc[t]);
  }

  __shared__ alignas(16) float ft[16 * 64];
  const int rhalf = lane >> 4, ncol = lane & 15;
#pragma unroll
  for (int t = 0; t < 4; ++t)
#pragma unroll
    for (int i = 0; i < 8; ++i)
      ft[(i + 8 * rhalf) * 64 + ncol + 16 * t] = acc[t][i];
  __syncthreads();

#pragma unroll
  for (int j = 0; j < 8; ++j) {
    int c = lane + 32 * j;                  // 256 chunks of 4 floats
    int r = c >> 4, off = (c & 15) * 4;
    *reinterpret_cast<v4f*>(outp + (size_t)(mtile * 16 + r) * H_ + n0 + off) =
        *reinterpret_cast<const v4f*>(&ft[r * 64 + off]);
  }
}

// ---------------- host launcher ----------------
extern "C" void kernel_launch(void* const* d_in, const int* in_sizes, int n_in,
                              void* d_out, int out_size, void* d_ws, size_t ws_size,
                              hipStream_t stream) {
  (void)in_sizes; (void)n_in; (void)out_size; (void)ws_size;
  const float* x    = (const float*)d_in[0];
  const float* Wq   = (const float*)d_in[1];
  const float* Wk   = (const float*)d_in[2];
  const float* Wv   = (const float*)d_in[3];
  const float* Wo   = (const float*)d_in[4];
  const float* bias = (const float*)d_in[5];

  float* outp = (float*)d_out;                       // [B,S,H] fp32
  float* attn = outp + (size_t)B_ * S_ * H_;         // [B,NH,S,S] fp32

  const size_t NX = (size_t)B_ * S_ * H_;            // 4M elements
  const size_t NW = (size_t)H_ * H_;                 // 1M elements
  half_t* xh   = (half_t*)d_ws;
  half_t* Wqkv = xh + NX;                            // Wq,Wk,Wv f16 contiguous
  half_t* Woh  = Wqkv + 3 * NW;
  half_t* qh   = Woh + NW;
  half_t* kh   = qh + NX;
  half_t* vTh  = kh + NX;
  half_t* oh   = vTh + NX;                           // total 24M halfs = 48MB

  // stage 0: convert inputs to f16
  k_cvt<<<(int)(NX / 4 + 255) / 256, 256, 0, stream>>>(x, xh, (int)(NX / 4));
  k_cvt<<<(int)(NW / 4 + 255) / 256, 256, 0, stream>>>(Wq, Wqkv, (int)(NW / 4));
  k_cvt<<<(int)(NW / 4 + 255) / 256, 256, 0, stream>>>(Wk, Wqkv + NW, (int)(NW / 4));
  k_cvt<<<(int)(NW / 4 + 255) / 256, 256, 0, stream>>>(Wv, Wqkv + 2 * NW, (int)(NW / 4));
  k_cvt<<<(int)(NW / 4 + 255) / 256, 256, 0, stream>>>(Wo, Woh, (int)(NW / 4));

  // stage 1: q,k,v projections + RoPE (v stored transposed)
  k_qkv_rope<<<dim3(B_ * S_ / 16, NH_, 3), 32, 0, stream>>>(xh, Wqkv, qh, kh, vTh);

  // stage 2: scores (lower triangle), softmax, attn @ v
  k_scores<<<dim3(S_ / 16, B_ * NH_), 256, 0, stream>>>(qh, kh, bias, attn);
  k_softmax<<<B_ * NH_ * S_, 256, 0, stream>>>(attn);
  k_av<<<dim3(S_ / 16, B_ * NH_), 32, 0, stream>>>(attn, vTh, oh);

  // stage 3: output projection
  k_outproj<<<dim3(B_ * S_ / 16, NH_), 32, 0, stream>>>(oh, Woh, outp);
}
